// Sub2GeneDifferCrossMHA_26903675142742
// MI455X (gfx1250) — matrix operand order repaired and checked by reading, not compiled, not run on hardware
//
#include <hip/hip_runtime.h>
#include <hip/hip_bf16.h>

// ---------------------------------------------------------------------------
// Sub2GeneDifferCrossMHA for MI455X (gfx1250, wave32, WMMA)
// ---------------------------------------------------------------------------
typedef __attribute__((ext_vector_type(16))) __bf16 v16bf;
typedef __attribute__((ext_vector_type(8)))  __bf16 v8bf;
typedef __attribute__((ext_vector_type(8)))  float  v8f;
typedef __attribute__((ext_vector_type(2)))  float  v2f;

#define NB   4
#define NS   512
#define NH   8
#define NHD  16
#define NLK  2048
#define ND2  256            // 2*H*HD
#define LAMBDA_INIT 0.35550906759096926f   // 0.8 - 0.6*exp(-0.3)

// optional gfx1250 async global->LDS path (guarded; sync fallback otherwise)
#if defined(__has_builtin)
#if __has_builtin(__builtin_amdgcn_global_load_async_to_lds_b128) && \
    __has_builtin(__builtin_amdgcn_s_wait_asynccnt)
#define HAVE_ASYNC_LDS 1
#endif
#endif
#ifdef HAVE_ASYNC_LDS
typedef int v4i __attribute__((ext_vector_type(4)));
typedef __attribute__((address_space(1))) v4i gas_v4i;   // printed as "__device__" AS
typedef __attribute__((address_space(3))) v4i las_v4i;   // printed as "__shared__" AS
#endif

// ---- workspace layout (bytes) ----
#define OFF_QBF   ((size_t)0)                        // query bf16  [B*S,256]
#define OFF_KBF   (OFF_QBF + (size_t)NB*NS*256*2)    // key bf16    [B*Lk,256]
#define OFF_WQT   (OFF_KBF + (size_t)NB*NLK*256*2)   // Wq^T bf16   [256,256]
#define OFF_WKT   (OFF_WQT + (size_t)256*256*2)
#define OFF_WVT   (OFF_WKT + (size_t)256*256*2)
#define OFF_WOT   (OFF_WVT + (size_t)256*256*2)
#define OFF_QP    (OFF_WOT + (size_t)256*256*2)      // Q proj f32 (scaled)
#define OFF_KP    (OFF_QP  + (size_t)NB*NS*256*4)    // K proj f32
#define OFF_VT    (OFF_KP  + (size_t)NB*NLK*256*4)   // V^T bf16 [B,H,32,Lk]
#define OFF_CTX   (OFF_VT  + (size_t)NB*NLK*256*2)   // ctx bf16 [B,S,256]
#define OFF_M0    (OFF_CTX + (size_t)NB*NS*256*2)    // softmax stats 4x [B,H,S] f32
#define OFF_S0    (OFF_M0  + (size_t)NB*NH*NS*4)
#define OFF_M1    (OFF_S0  + (size_t)NB*NH*NS*4)
#define OFF_S1    (OFF_M1  + (size_t)NB*NH*NS*4)
#define OFF_LAM   (OFF_S1  + (size_t)NB*NH*NS*4)

__device__ __forceinline__ float groupSum16(float v) {
#pragma unroll
  for (int off = 1; off < 16; off <<= 1) v += __shfl_xor(v, off, 32);
  return v;
}

// ---------------------------------------------------------------------------
// conversion / setup kernels
// ---------------------------------------------------------------------------
__global__ void k_f2bf(const float* __restrict__ src, __bf16* __restrict__ dst, int n) {
  int i = blockIdx.x * blockDim.x + threadIdx.x;
  if (i < n) dst[i] = (__bf16)src[i];
}
__global__ void k_wt(const float* __restrict__ w, __bf16* __restrict__ wt) {
  int i = blockIdx.x * blockDim.x + threadIdx.x;  // 65536
  int k = i >> 8, n = i & 255;
  wt[n * 256 + k] = (__bf16)w[i];
}
__global__ void k_lambda(const float* lq1, const float* lk1,
                         const float* lq2, const float* lk2, float* lam) {
  if (threadIdx.x == 0 && blockIdx.x == 0) {
    float s1 = 0.f, s2 = 0.f;
    for (int i = 0; i < NHD; ++i) { s1 += lq1[i] * lk1[i]; s2 += lq2[i] * lk2[i]; }
    *lam = expf(s1) - expf(s2) + LAMBDA_INIT;
  }
}

// ---------------------------------------------------------------------------
// bf16 GEMM: C[M,256] = A[M,256] @ W[256,256], W given transposed [N][K].
// 128 threads = 4 waves; block covers a 16x64 tile of C. The 16x256 A tile is
// staged to LDS once per block (async global->LDS when available), then each
// wave reads its WMMA A-fragments from LDS and B-fragments from global Wt.
// MODE 0: Q proj -> f32*0.25   MODE 1: K proj -> f32
// MODE 2: V proj -> bf16 scattered to Vt[B,H,32,Lk]   MODE 3: out proj -> f32
// ---------------------------------------------------------------------------
template <int MODE>
__global__ __launch_bounds__(128) void k_gemm(const __bf16* __restrict__ A,
                                              const __bf16* __restrict__ Wt,
                                              float* __restrict__ fout,
                                              __bf16* __restrict__ bfout) {
  __shared__ __attribute__((aligned(16))) __bf16 atile[16 * 256];
  const int lane = threadIdx.x & 31;
  const int wid  = threadIdx.x >> 5;
  const int half = lane >> 4;
  const int r16  = lane & 15;
  const int m0   = (blockIdx.x >> 2) * 16;
  const int n0   = (blockIdx.x & 3) * 64 + wid * 16;

  // ---- stage A tile (16 rows x 256 halves = 8 KB) into LDS ----
#pragma unroll
  for (int c = 0; c < 4; ++c) {
    const int chunk = threadIdx.x + c * 128;   // 512 chunks of 8 halves (16B)
    const int row = chunk >> 5;
    const int off = (chunk & 31) * 8;
    const __bf16* gsrc = A + (size_t)(m0 + row) * 256 + off;
    __bf16* ldst = &atile[row * 256 + off];
#ifdef HAVE_ASYNC_LDS
    __builtin_amdgcn_global_load_async_to_lds_b128((gas_v4i*)gsrc, (las_v4i*)ldst, 0, 0);
#else
    *(v8bf*)ldst = *(const v8bf*)gsrc;
#endif
  }
#ifdef HAVE_ASYNC_LDS
  __builtin_amdgcn_s_wait_asynccnt(0);
#endif
  __syncthreads();

  const __bf16* arow = atile + r16 * 256;                         // A: lane row
  const __bf16* brow = Wt + (size_t)(n0 + r16) * 256 + half * 16; // B: lane col
  v8f acc = {};
#pragma unroll
  for (int k0 = 0; k0 < 256; k0 += 32) {
    v8bf alo = *(const v8bf*)(arow + k0 + half * 16);       // K {0..7}/{16..23}
    v8bf ahi = *(const v8bf*)(arow + k0 + half * 16 + 8);   // K {8..15}/{24..31}
    v16bf av = __builtin_shufflevector(alo, ahi, 0,1,2,3,4,5,6,7,8,9,10,11,12,13,14,15);
    v16bf bv = *(const v16bf*)(brow + k0);                  // 16 consecutive K
    acc = __builtin_amdgcn_wmma_f32_16x16x32_bf16(false, av, false, bv,
                                                  (short)0, acc, false, false);
  }
  const int col = n0 + r16;   // D layout: lane col, rows m0 + half*8 + j
#pragma unroll
  for (int j = 0; j < 8; ++j) {
    const int gm = m0 + half * 8 + j;
    float v = acc[j];
    if (MODE == 0) fout[(size_t)gm * 256 + col] = v * 0.25f;   // * HD^-0.5
    if (MODE == 1) fout[(size_t)gm * 256 + col] = v;
    if (MODE == 3) fout[(size_t)gm * 256 + col] = v;
    if (MODE == 2) {
      const int b = gm >> 11, l = gm & (NLK - 1);
      const int h = col >> 5, dd = col & 31;
      bfout[(((size_t)b * NH + h) * 32 + dd) * NLK + l] = (__bf16)v;
    }
  }
}

// ---------------------------------------------------------------------------
// Pass 1: softmax stats (max, sum-exp) for both slots.
// f32 WMMA 16x16x4 chained over HD=16 (exact fp32 logits).
// Each lane owns one score COLUMN of the C tile, so it runs a lane-local
// online softmax (pure VALU) over all its columns; the 16-lane (m,s) merge
// happens once at the end -> no cross-lane traffic in the hot loop.
// ---------------------------------------------------------------------------
__global__ __launch_bounds__(128) void k_attn_stats(
    const float* __restrict__ Qp, const float* __restrict__ Kp,
    const unsigned char* __restrict__ qmask, const unsigned char* __restrict__ kmask,
    float* __restrict__ stm0, float* __restrict__ sts0,
    float* __restrict__ stm1, float* __restrict__ sts1) {
  const int lane = threadIdx.x & 31;
  const int wid  = threadIdx.x >> 5;
  const int half = lane >> 4;
  const int r16  = lane & 15;
  const int b  = blockIdx.x >> 6;
  const int h  = (blockIdx.x >> 3) & 7;
  const int s0 = (blockIdx.x & 7) * 64 + wid * 16;

  const float* qrow = Qp + ((size_t)(b * NS + s0 + r16)) * 256 + h * 32;
  v2f qa[2][4];
#pragma unroll
  for (int t = 0; t < 2; ++t)
#pragma unroll
    for (int kc = 0; kc < 4; ++kc)
      qa[t][kc] = *(const v2f*)(qrow + t * 16 + kc * 4 + half * 2);

  bool qm[8];
#pragma unroll
  for (int j = 0; j < 8; ++j) qm[j] = qmask[b * NS + s0 + half * 8 + j] != 0;

  float m[2][8], ss[2][8];
#pragma unroll
  for (int t = 0; t < 2; ++t)
#pragma unroll
    for (int j = 0; j < 8; ++j) { m[t][j] = -INFINITY; ss[t][j] = 0.f; }

  const float* kbase = Kp + (size_t)b * NLK * 256 + h * 32;
  for (int l0 = 0; l0 < NLK; l0 += 16) {
    const bool km = kmask[b * NLK + l0 + r16] != 0;
    const float* krow = kbase + (size_t)(l0 + r16) * 256;
    __builtin_prefetch(krow + 16 * 256, 0, 1);   // global_prefetch_b8
#pragma unroll
    for (int t = 0; t < 2; ++t) {
      v8f acc = {};
#pragma unroll
      for (int kc = 0; kc < 4; ++kc) {
        v2f bv = *(const v2f*)(krow + t * 16 + kc * 4 + half * 2);
        acc = __builtin_amdgcn_wmma_f32_16x16x4_f32(false, qa[t][kc], false, bv,
                                                    (short)0, acc, false, false);
      }
#pragma unroll
      for (int j = 0; j < 8; ++j) {              // lane-local online update
        float x  = (km && qm[j]) ? acc[j] : -1e20f;
        float mn = fmaxf(m[t][j], x);
        ss[t][j] = ss[t][j] * expf(m[t][j] - mn) + expf(x - mn);
        m[t][j]  = mn;
      }
    }
  }
  // merge (m,s) across the 16 lanes of each half (4 butterfly steps, once)
#pragma unroll
  for (int t = 0; t < 2; ++t)
#pragma unroll
    for (int j = 0; j < 8; ++j) {
#pragma unroll
      for (int off = 1; off < 16; off <<= 1) {
        float mo = __shfl_xor(m[t][j], off, 32);
        float so = __shfl_xor(ss[t][j], off, 32);
        float mn = fmaxf(m[t][j], mo);
        ss[t][j] = ss[t][j] * expf(m[t][j] - mn) + so * expf(mo - mn);
        m[t][j]  = mn;
      }
    }
  if (r16 == 0) {
    const int base = (b * NH + h) * NS + s0 + half * 8;
#pragma unroll
    for (int j = 0; j < 8; ++j) {
      stm0[base + j] = m[0][j]; sts0[base + j] = ss[0][j];
      stm1[base + j] = m[1][j]; sts1[base + j] = ss[1][j];
    }
  }
}

// ---------------------------------------------------------------------------
// Pass 2: recompute logits, emit diff_attn to d_out, accumulate diff @ V_cat
// with bf16 WMMA (diff tile staged through LDS to transpose C->A layout),
// RMSNorm epilogue -> ctx bf16.
// ---------------------------------------------------------------------------
__global__ __launch_bounds__(128) void k_attn_out(
    const float* __restrict__ Qp, const float* __restrict__ Kp,
    const __bf16* __restrict__ Vt,
    const unsigned char* __restrict__ qmask, const unsigned char* __restrict__ kmask,
    const float* __restrict__ stm0, const float* __restrict__ sts0,
    const float* __restrict__ stm1, const float* __restrict__ sts1,
    const float* __restrict__ lamp, const float* __restrict__ g,
    float* __restrict__ diff_out, __bf16* __restrict__ ctx) {
  // per-wave 16x32 bf16 staging tile, stride 40 halves (16B-aligned chunks)
  __shared__ __attribute__((aligned(32))) __bf16 stage[4][16 * 40];
  const int lane = threadIdx.x & 31;
  const int wid  = threadIdx.x >> 5;
  const int half = lane >> 4;
  const int r16  = lane & 15;
  const int b  = blockIdx.x >> 6;
  const int h  = (blockIdx.x >> 3) & 7;
  const int s0 = (blockIdx.x & 7) * 64 + wid * 16;

  const float* qrow = Qp + ((size_t)(b * NS + s0 + r16)) * 256 + h * 32;
  v2f qa[2][4];
#pragma unroll
  for (int t = 0; t < 2; ++t)
#pragma unroll
    for (int kc = 0; kc < 4; ++kc)
      qa[t][kc] = *(const v2f*)(qrow + t * 16 + kc * 4 + half * 2);

  bool qm[8];
#pragma unroll
  for (int j = 0; j < 8; ++j) qm[j] = qmask[b * NS + s0 + half * 8 + j] != 0;

  float m0_[8], is0[8], m1_[8], is1[8];
  {
    const int base = (b * NH + h) * NS + s0 + half * 8;
#pragma unroll
    for (int j = 0; j < 8; ++j) {
      m0_[j] = stm0[base + j]; is0[j] = 1.f / (sts0[base + j] + 1e-20f);
      m1_[j] = stm1[base + j]; is1[j] = 1.f / (sts1[base + j] + 1e-20f);
    }
  }
  const float lam = *lamp;
  const float g0 = g[r16], g1 = g[16 + r16];

  const float* kbase = Kp + (size_t)b * NLK * 256 + h * 32;
  float* dbase = diff_out + (size_t)(b * NH + h) * NS * NLK;   // [B,H,S,Lk]
  __bf16* st = &stage[wid][0];

  v8f acc0 = {}, acc1 = {};
  for (int l0 = 0; l0 < NLK; l0 += 32) {
#pragma unroll
    for (int sub = 0; sub < 2; ++sub) {
      const int lc = l0 + sub * 16;
      const bool km = kmask[b * NLK + lc + r16] != 0;
      const float* krow = kbase + (size_t)(lc + r16) * 256;
      __builtin_prefetch(krow + 16 * 256, 0, 1);
      v8f x0 = {}, x1 = {};
#pragma unroll
      for (int kc = 0; kc < 4; ++kc) {
        v2f bv0 = *(const v2f*)(krow + kc * 4 + half * 2);
        v2f bv1 = *(const v2f*)(krow + 16 + kc * 4 + half * 2);
        x0 = __builtin_amdgcn_wmma_f32_16x16x4_f32(false, qa[0][kc], false, bv0,
                                                   (short)0, x0, false, false);
        x1 = __builtin_amdgcn_wmma_f32_16x16x4_f32(false, qa[1][kc], false, bv1,
                                                   (short)0, x1, false, false);
      }
#pragma unroll
      for (int j = 0; j < 8; ++j) {
        const bool ok = km && qm[j];
        float a0 = ok ? x0[j] : -1e20f;
        float a1 = ok ? x1[j] : -1e20f;
        float p0 = expf(a0 - m0_[j]) * is0[j];
        float p1 = expf(a1 - m1_[j]) * is1[j];
        float d  = p0 - lam * p1;
        dbase[(size_t)(s0 + half * 8 + j) * NLK + lc + r16] = d;
        st[(half * 8 + j) * 40 + sub * 16 + r16] = (__bf16)d;
      }
    }
    // A fragment (16x32 bf16) from staged diff tile; B fragments from Vt rows
    v8bf alo = *(const v8bf*)(st + r16 * 40 + half * 16);
    v8bf ahi = *(const v8bf*)(st + r16 * 40 + half * 16 + 8);
    v16bf av = __builtin_shufflevector(alo, ahi, 0,1,2,3,4,5,6,7,8,9,10,11,12,13,14,15);
    const __bf16* vrow = Vt + (((size_t)b * NH + h) * 32 + r16) * NLK + l0 + half * 16;
    v16bf bv0 = *(const v16bf*)(vrow);                // dims 0..15
    v16bf bv1 = *(const v16bf*)(vrow + 16 * NLK);     // dims 16..31
    acc0 = __builtin_amdgcn_wmma_f32_16x16x32_bf16(false, av, false, bv0,
                                                   (short)0, acc0, false, false);
    acc1 = __builtin_amdgcn_wmma_f32_16x16x32_bf16(false, av, false, bv1,
                                                   (short)0, acc1, false, false);
  }
  // RMSNorm over 32 dims per row, scale by g and (1 - lambda_init)
#pragma unroll
  for (int j = 0; j < 8; ++j) {
    float sumsq = groupSum16(acc0[j] * acc0[j] + acc1[j] * acc1[j]);
    float sc = rsqrtf(sumsq * (1.0f / 32.0f) + 1e-5f) * (1.0f - LAMBDA_INIT);
    const int s = s0 + half * 8 + j;
    __bf16* cp = ctx + ((size_t)b * NS + s) * 256 + h * 32;
    cp[r16]      = (__bf16)(acc0[j] * sc * g0);
    cp[16 + r16] = (__bf16)(acc1[j] * sc * g1);
  }
}

// ---------------------------------------------------------------------------
extern "C" void kernel_launch(void* const* d_in, const int* in_sizes, int n_in,
                              void* d_out, int out_size, void* d_ws, size_t ws_size,
                              hipStream_t stream) {
  const float* query = (const float*)d_in[0];
  const float* key   = (const float*)d_in[1];
  const unsigned char* qmask = (const unsigned char*)d_in[2];
  const unsigned char* kmask = (const unsigned char*)d_in[3];
  const float* Wq = (const float*)d_in[4];
  const float* Wk = (const float*)d_in[5];
  const float* Wv = (const float*)d_in[6];
  const float* Wo = (const float*)d_in[7];
  const float* lq1 = (const float*)d_in[8];
  const float* lk1 = (const float*)d_in[9];
  const float* lq2 = (const float*)d_in[10];
  const float* lk2 = (const float*)d_in[11];
  const float* g   = (const float*)d_in[12];

  char* ws = (char*)d_ws;
  __bf16* qbf = (__bf16*)(ws + OFF_QBF);
  __bf16* kbf = (__bf16*)(ws + OFF_KBF);
  __bf16* wqt = (__bf16*)(ws + OFF_WQT);
  __bf16* wkt = (__bf16*)(ws + OFF_WKT);
  __bf16* wvt = (__bf16*)(ws + OFF_WVT);
  __bf16* wot = (__bf16*)(ws + OFF_WOT);
  float*  Qp  = (float*)(ws + OFF_QP);
  float*  Kp  = (float*)(ws + OFF_KP);
  __bf16* Vt  = (__bf16*)(ws + OFF_VT);
  __bf16* ctx = (__bf16*)(ws + OFF_CTX);
  float*  stm0 = (float*)(ws + OFF_M0);
  float*  sts0 = (float*)(ws + OFF_S0);
  float*  stm1 = (float*)(ws + OFF_M1);
  float*  sts1 = (float*)(ws + OFF_S1);
  float*  lam  = (float*)(ws + OFF_LAM);

  float* out_proj = (float*)d_out;                         // [B,S,256]
  float* diff_out = (float*)d_out + (size_t)NB * NS * 256; // [B,H,S,Lk]

  const int nq = NB * NS * 256, nk = NB * NLK * 256;
  k_f2bf<<<(nq + 255) / 256, 256, 0, stream>>>(query, qbf, nq);
  k_f2bf<<<(nk + 255) / 256, 256, 0, stream>>>(key, kbf, nk);
  k_wt<<<256, 256, 0, stream>>>(Wq, wqt);
  k_wt<<<256, 256, 0, stream>>>(Wk, wkt);
  k_wt<<<256, 256, 0, stream>>>(Wv, wvt);
  k_wt<<<256, 256, 0, stream>>>(Wo, wot);
  k_lambda<<<1, 32, 0, stream>>>(lq1, lk1, lq2, lk2, lam);

  k_gemm<0><<<(NB * NS / 16) * 4, 128, 0, stream>>>(qbf, wqt, Qp, nullptr);
  k_gemm<1><<<(NB * NLK / 16) * 4, 128, 0, stream>>>(kbf, wkt, Kp, nullptr);
  k_gemm<2><<<(NB * NLK / 16) * 4, 128, 0, stream>>>(kbf, wvt, nullptr, Vt);

  k_attn_stats<<<NB * NH * (NS / 64), 128, 0, stream>>>(Qp, Kp, qmask, kmask,
                                                        stm0, sts0, stm1, sts1);
  k_attn_out<<<NB * NH * (NS / 64), 128, 0, stream>>>(Qp, Kp, Vt, qmask, kmask,
                                                      stm0, sts0, stm1, sts1,
                                                      lam, g, diff_out, ctx);

  k_gemm<3><<<(NB * NS / 16) * 4, 128, 0, stream>>>(ctx, wot, out_proj, nullptr);
}
